// DecSwitchedFC_34187939676856
// MI455X (gfx1250) — matrix operand order repaired and checked by reading, not compiled, not run on hardware
//
#include <hip/hip_runtime.h>

// ---------------- types ----------------
typedef __attribute__((ext_vector_type(16))) __bf16 v16bf;
typedef __attribute__((ext_vector_type(8)))  float  v8f;
typedef __attribute__((ext_vector_type(4)))  unsigned int u32x4;
typedef __attribute__((ext_vector_type(4)))  float  f32x4;

#define NDIMS    1024
#define NSM      256
#define NBR      16
#define BATCH    8192
#define MTILE    32          // rows per block
#define KCHUNK   512         // x-tile K staging chunk
#define A_STRIDE (KCHUNK + 8)  // 520 bf16 -> row stride 260 dwords -> 4-bank rotation
#define H_STRIDE (NSM + 8)     // 264 bf16 -> row stride 132 dwords -> 4-bank rotation

__device__ __forceinline__ unsigned short f2bf(float f) {
    union { float f; unsigned u; } v; v.f = f;
    unsigned u = v.u;
    unsigned r = ((u >> 16) & 1u) + 0x7FFFu;   // round-to-nearest-even
    return (unsigned short)((u + r) >> 16);
}

// ---------------- kernel 1: zero branch counters ----------------
__global__ void zero_counts(int* counts) {
    if (threadIdx.x < NBR) counts[threadIdx.x] = 0;
}

// ---------------- kernel 2: bin rows by branch ----------------
__global__ void bin_rows(const int* __restrict__ y_idx,
                         int* __restrict__ counts,
                         int* __restrict__ rowlist) {
    int b = blockIdx.x * blockDim.x + threadIdx.x;
    if (b < BATCH) {
        int n = y_idx[b];
        int pos = atomicAdd(&counts[n], 1);
        rowlist[n * BATCH + pos] = b;
    }
}

// ---------------- kernel 3/4: transpose + f32->bf16 convert ----------------
// src: [NBR][R][C] f32 -> dst: [NBR][C][R] bf16 (weight rows become K-contiguous)
__global__ void transpose_convert(const float* __restrict__ src,
                                  unsigned short* __restrict__ dst,
                                  int R, int C) {
    __shared__ float tile[32][33];
    int n  = blockIdx.z;
    int r0 = blockIdx.x * 32, c0 = blockIdx.y * 32;
    const float* s = src + (size_t)n * R * C;
    unsigned short* d = dst + (size_t)n * R * C;
    int tx = threadIdx.x, ty = threadIdx.y;   // block (32, 8)
    for (int i = 0; i < 32; i += 8)
        tile[ty + i][tx] = s[(size_t)(r0 + ty + i) * C + (c0 + tx)];
    __syncthreads();
    for (int i = 0; i < 32; i += 8)
        d[(size_t)(c0 + ty + i) * R + (r0 + tx)] = f2bf(tile[tx][ty + i]);
}

// ---------------- kernel 5: grouped bottleneck FC via bf16 WMMA ----------------
// grid = (NBR branches, BATCH/MTILE tiles); block = 256 threads (8 wave32)
// Each block: 32 rows of one branch. FC1: per wave 2 M-tiles x 2 N-tiles.
// FC2: two passes, per pass 2 M-tiles x 4 N-tiles. Every B fragment feeds 2 WMMAs.
__global__ __launch_bounds__(256)
void switched_fc_wmma(const float* __restrict__ x,
                      const float* __restrict__ z,
                      const float* __restrict__ b1,
                      const float* __restrict__ b2,
                      const int*   __restrict__ counts,
                      const int*   __restrict__ rowlist,
                      const unsigned short* __restrict__ W1t,  // [NBR][NSM][NDIMS] bf16
                      const unsigned short* __restrict__ W2t,  // [NBR][NDIMS][NSM] bf16
                      float* __restrict__ out) {
    __shared__ unsigned short Alds[MTILE * A_STRIDE];  // 33,280 B: x K-chunk, bf16
    __shared__ unsigned short Hlds[MTILE * H_STRIDE];  // 16,896 B: hidden tile, bf16
    __shared__ int   rows_s[MTILE];
    __shared__ float z_s[MTILE];
    __shared__ int   valid_s;

    const int n    = blockIdx.x;
    const int base = blockIdx.y * MTILE;
    const int cnt  = counts[n];
    if (base >= cnt) return;               // whole block exits: no EXEC partials

    const int tid = threadIdx.x;
    if (tid < MTILE) {
        int idx = base + tid;
        int row = (idx < cnt) ? rowlist[n * BATCH + idx] : rowlist[n * BATCH + base];
        rows_s[tid] = row;
        z_s[tid]    = z[row * NBR + n];
        if (tid == 0) valid_s = (cnt - base < MTILE) ? (cnt - base) : MTILE;
    }

    const int lane = tid & 31;
    const int wv   = tid >> 5;
    const int half = lane >> 4;     // 0: lanes 0-15, 1: lanes 16-31
    const int lr   = lane & 15;

    // ---------------- FC1: h = relu(x @ W1 + b1) --------------------------
    const unsigned short* W1n = W1t + (size_t)n * NSM * NDIMS;
    const int nt0 = wv * 2, nt1 = wv * 2 + 1;
    v8f za = {0.f,0.f,0.f,0.f,0.f,0.f,0.f,0.f};
    v8f accA[2][2] = {{za, za}, {za, za}};   // [mtile][ntile]

    for (int kc = 0; kc < NDIMS; kc += KCHUNK) {
        __syncthreads();   // previous chunk consumed (first iter: rows_s ready)
        // Stage 32 x KCHUNK x-slab into LDS as bf16 (float4 loads, coalesced)
        for (int i = tid; i < MTILE * (KCHUNK / 4); i += 256) {
            int r  = i >> 7;           // KCHUNK/4 == 128
            int c4 = i & 127;
            f32x4 v = *(const f32x4*)(x + (size_t)rows_s[r] * NDIMS + kc + c4 * 4);
            unsigned short* p = &Alds[r * A_STRIDE + c4 * 4];
            p[0] = f2bf(v.x); p[1] = f2bf(v.y); p[2] = f2bf(v.z); p[3] = f2bf(v.w);
        }
        __syncthreads();

        const unsigned short* bp0 = W1n + (size_t)(nt0 * 16 + lr) * NDIMS + kc;
        const unsigned short* bp1 = W1n + (size_t)(nt1 * 16 + lr) * NDIMS + kc;
        for (int k0 = 0; k0 < KCHUNK; k0 += 32) {
            // A 16x32 bf16 fragment per M-tile:
            //   lane-half 0 -> K {k0..k0+7, k0+16..k0+23}; half 1 -> {+8.., +24..}
            union { v16bf v; u32x4 q[2]; } af0, af1, bf0, bf1;
            const int kA = k0 + half * 8;
            af0.q[0] = *(const u32x4*)(&Alds[lr * A_STRIDE + kA]);
            af0.q[1] = *(const u32x4*)(&Alds[lr * A_STRIDE + kA + 16]);
            af1.q[0] = *(const u32x4*)(&Alds[(16 + lr) * A_STRIDE + kA]);
            af1.q[1] = *(const u32x4*)(&Alds[(16 + lr) * A_STRIDE + kA + 16]);
            // B 32x16 fragment: lanes 0-15 hold K {k0..k0+7, k0+16..k0+23}, 16-31 {+8.., +24..}
            const int kB = k0 + half * 8;
            bf0.q[0] = *(const u32x4*)(bp0 + kB);
            bf0.q[1] = *(const u32x4*)(bp0 + kB + 16);
            bf1.q[0] = *(const u32x4*)(bp1 + kB);
            bf1.q[1] = *(const u32x4*)(bp1 + kB + 16);
            __builtin_prefetch(bp0 + kB + 32, 0, 3);   // warm next K-step
            __builtin_prefetch(bp1 + kB + 32, 0, 3);
            accA[0][0] = __builtin_amdgcn_wmma_f32_16x16x32_bf16(false, af0.v, false, bf0.v,
                                                                 (short)0, accA[0][0], false, false);
            accA[1][0] = __builtin_amdgcn_wmma_f32_16x16x32_bf16(false, af1.v, false, bf0.v,
                                                                 (short)0, accA[1][0], false, false);
            accA[0][1] = __builtin_amdgcn_wmma_f32_16x16x32_bf16(false, af0.v, false, bf1.v,
                                                                 (short)0, accA[0][1], false, false);
            accA[1][1] = __builtin_amdgcn_wmma_f32_16x16x32_bf16(false, af1.v, false, bf1.v,
                                                                 (short)0, accA[1][1], false, false);
        }
    }

    // bias + relu, write h (C/D layout: VGPR r, lane-half h -> M = r + 8*h, N = lr)
    {
        float bias0 = b1[n * NSM + nt0 * 16 + lr];
        float bias1 = b1[n * NSM + nt1 * 16 + lr];
        __syncthreads();   // Alds reads done before any wave races ahead
#pragma unroll
        for (int mt = 0; mt < 2; ++mt) {
#pragma unroll
            for (int r = 0; r < 8; ++r) {
                int m = mt * 16 + r + half * 8;
                float v0 = accA[mt][0][r] + bias0; v0 = v0 > 0.f ? v0 : 0.f;
                float v1 = accA[mt][1][r] + bias1; v1 = v1 > 0.f ? v1 : 0.f;
                Hlds[m * H_STRIDE + nt0 * 16 + lr] = f2bf(v0);
                Hlds[m * H_STRIDE + nt1 * 16 + lr] = f2bf(v1);
            }
        }
    }
    __syncthreads();

    // ---------------- FC2: o = h @ W2 + b2; out = x + z * o ----------------
    const unsigned short* W2n = W2t + (size_t)n * NDIMS * NSM;
    const int valid = valid_s;

#pragma unroll
    for (int p = 0; p < 2; ++p) {          // two N-passes of 512 columns
        v8f acc[2][4];                     // [mtile][ntile]
#pragma unroll
        for (int mt = 0; mt < 2; ++mt)
#pragma unroll
            for (int j = 0; j < 4; ++j)
                acc[mt][j] = (v8f){0.f,0.f,0.f,0.f,0.f,0.f,0.f,0.f};

        for (int k0 = 0; k0 < NSM; k0 += 32) {
            union { v16bf v; u32x4 q[2]; } af0, af1;
            const int kA = k0 + half * 8;
            af0.q[0] = *(const u32x4*)(&Hlds[lr * H_STRIDE + kA]);
            af0.q[1] = *(const u32x4*)(&Hlds[lr * H_STRIDE + kA + 16]);
            af1.q[0] = *(const u32x4*)(&Hlds[(16 + lr) * H_STRIDE + kA]);
            af1.q[1] = *(const u32x4*)(&Hlds[(16 + lr) * H_STRIDE + kA + 16]);
            const int kB = k0 + half * 8;
#pragma unroll
            for (int j = 0; j < 4; ++j) {
                int nc = (p * 32 + wv * 4 + j) * 16 + lr;
                union { v16bf v; u32x4 q[2]; } bfj;
                const unsigned short* bp = W2n + (size_t)nc * NSM + kB;
                bfj.q[0] = *(const u32x4*)(bp);
                bfj.q[1] = *(const u32x4*)(bp + 16);
                acc[0][j] = __builtin_amdgcn_wmma_f32_16x16x32_bf16(false, af0.v, false, bfj.v,
                                                                    (short)0, acc[0][j], false, false);
                acc[1][j] = __builtin_amdgcn_wmma_f32_16x16x32_bf16(false, af1.v, false, bfj.v,
                                                                    (short)0, acc[1][j], false, false);
            }
        }

        // epilogue: out[row] = x[row] + z[row] * (o + b2), mask padded rows
#pragma unroll
        for (int j = 0; j < 4; ++j) {
            int col = (p * 32 + wv * 4 + j) * 16 + lr;
            float bias = b2[n * NDIMS + col];
#pragma unroll
            for (int mt = 0; mt < 2; ++mt) {
#pragma unroll
                for (int r = 0; r < 8; ++r) {
                    int m = mt * 16 + r + half * 8;
                    if (m < valid) {
                        int row = rows_s[m];
                        float o = acc[mt][j][r] + bias;
                        out[(size_t)row * NDIMS + col] =
                            x[(size_t)row * NDIMS + col] + z_s[m] * o;
                    }
                }
            }
        }
    }
}

// ---------------- launch ----------------
extern "C" void kernel_launch(void* const* d_in, const int* in_sizes, int n_in,
                              void* d_out, int out_size, void* d_ws, size_t ws_size,
                              hipStream_t stream) {
    const float* x   = (const float*)d_in[0];
    const int* y_idx = (const int*)  d_in[1];
    // d_in[2] = y : unused by the reference output
    const float* z   = (const float*)d_in[3];
    const float* W1  = (const float*)d_in[4];
    const float* b1  = (const float*)d_in[5];
    const float* W2  = (const float*)d_in[6];
    const float* b2  = (const float*)d_in[7];
    float* out = (float*)d_out;
    (void)in_sizes; (void)n_in; (void)out_size; (void)ws_size;

    // workspace layout (~16.5 MB): counts | rowlist | W1t bf16 | W2t bf16
    char* ws = (char*)d_ws;
    int* counts  = (int*)ws;                                   // 64 B slot
    int* rowlist = (int*)(ws + 64);                            // 16*8192*4 = 512 KB
    unsigned short* W1t = (unsigned short*)(ws + 64 + sizeof(int) * NBR * BATCH);
    unsigned short* W2t = W1t + (size_t)NBR * NDIMS * NSM;

    zero_counts<<<1, 32, 0, stream>>>(counts);
    bin_rows<<<BATCH / 256, 256, 0, stream>>>(y_idx, counts, rowlist);
    transpose_convert<<<dim3(NDIMS / 32, NSM / 32, NBR), dim3(32, 8), 0, stream>>>(W1, W1t, NDIMS, NSM);
    transpose_convert<<<dim3(NSM / 32, NDIMS / 32, NBR), dim3(32, 8), 0, stream>>>(W2, W2t, NSM, NDIMS);
    switched_fc_wmma<<<dim3(NBR, BATCH / MTILE), 256, 0, stream>>>(
        x, z, b1, b2, counts, rowlist, W1t, W2t, out);
}
